// CapsNet_21277267984740
// MI455X (gfx1250) — matrix operand order repaired and checked by reading, not compiled, not run on hardware
//
#include <hip/hip_runtime.h>
#include <hip/hip_bf16.h>
#include <stdint.h>

// ---------------------------------------------------------------------------
// CapsNet forward on MI455X (gfx1250, wave32, WMMA).
//   convs + FCs : implicit-GEMM, v_wmma_f32_16x16x32_bf16 (bf16 in, f32 acc)
//                 128x64 block tile, 4 WMMA accumulators / wave,
//                 weight staging via GLOBAL_LOAD_ASYNC_TO_LDS_B128 when legal
//   u_hat       : batched 16x16x8 with v_wmma_f32_16x16x4_f32 (full fp32)
//   routing etc : bandwidth-bound VALU kernels (K-split reductions)
// ---------------------------------------------------------------------------

typedef __bf16 bf16;
typedef __attribute__((ext_vector_type(16))) __bf16 v16bf;
typedef __attribute__((ext_vector_type(8)))  float  v8f;
typedef __attribute__((ext_vector_type(2)))  float  v2f;

union FragB16 { v16bf v; uint4 q[2]; };

__device__ __forceinline__ float b2f(bf16 x) { return (float)x; }
__device__ __forceinline__ bf16  f2b(float x) { return (bf16)x; }

#if defined(__has_builtin)
#if __has_builtin(__builtin_amdgcn_global_load_async_to_lds_b128) && \
    __has_builtin(__builtin_amdgcn_s_wait_asynccnt)
#define HAVE_ASYNC_LDS 1
#endif
#endif
#ifndef HAVE_ASYNC_LDS
#define HAVE_ASYNC_LDS 0
#endif

#if HAVE_ASYNC_LDS
// param0: AS(1) int4*  (exact type per clang diagnostic), param1: AS(3) int4*
typedef int v4i_ __attribute__((vector_size(16)));
typedef __attribute__((address_space(1))) v4i_ gv4i;
typedef __attribute__((address_space(3))) v4i_ sv4i;
__device__ __forceinline__ void async_ld16(const void* g, void* l) {
    __builtin_amdgcn_global_load_async_to_lds_b128((gv4i*)g, (sv4i*)l, 0, 0);
}
#endif

// --------------------------- utility kernels -------------------------------

__global__ void k_cvt_bf16(const float* __restrict__ s, bf16* __restrict__ d, long n) {
    long i  = (long)blockIdx.x * blockDim.x + threadIdx.x;
    long st = (long)gridDim.x * blockDim.x;
    for (; i < n; i += st) d[i] = f2b(s[i]);
}

__global__ void k_zero_u32(uint32_t* __restrict__ d, long n) {
    long i  = (long)blockIdx.x * blockDim.x + threadIdx.x;
    long st = (long)gridDim.x * blockDim.x;
    for (; i < n; i += st) d[i] = 0u;
}

// --------------------- implicit-GEMM conv (bf16 WMMA) ----------------------
// GEMM view: C[M=B*OH*OW, N=OC] = A[M,K] * B[K,N],  K = IC*KH*KW (zero-padded
// to a multiple of 32).  Block tile 128x64, 8 waves in 4(M) x 2(N) layout,
// each wave owns 2 M-subtiles x 2 N-subtiles -> 4 WMMAs per K-chunk with
// A/B fragment reuse.  LDS stages A[128][32] and B^T[64][32] per chunk;
// the weight (B^T) stage is one contiguous 16B transfer per thread and uses
// the CDNA5 async-to-LDS path when K needs no zero padding.
__global__ __launch_bounds__(256) void k_conv_wmma(
    const bf16* __restrict__ in, const bf16* __restrict__ w,
    const float* __restrict__ bias, bf16* __restrict__ out,
    int B, int IC, int IHP, int IWP,
    int OC, int OH, int OW, int KH, int KW, int stride,
    int DH, int DW, int padOut, int Kreal, int relu)
{
    __shared__ __align__(16) bf16 As[128][32];
    __shared__ __align__(16) bf16 Bs[64][32];   // B^T: [n][k]

    const int M       = B * OH * OW;
    const int mBlocks = M >> 7;
    const int bm = blockIdx.x % mBlocks;
    const int bn = blockIdx.x / mBlocks;
    const int m0 = bm * 128, n0 = bn * 64;

    const int tid  = threadIdx.x;
    const int lane = tid & 31, wv = tid >> 5;
    const int lr = lane & 15, hi = lane >> 4;
    const int wm = wv & 3, wn = wv >> 2;
    const int KHW  = KH * KW;
    const int Kpad = (Kreal + 31) & ~31;
    const bool asyncB = HAVE_ASYNC_LDS && ((Kreal & 31) == 0);

    // A staging: thread t loads row t/2 (0..127), 16 cols at (t&1)*16
    const int ldrA  = tid >> 1;
    const int ldcA  = (tid & 1) * 16;
    const int am  = m0 + ldrA;
    const int aow = am % OW; const int t1 = am / OW;
    const int aoh = t1 % OH; const int ab = t1 / OH;
    const int ihb = aoh * stride, iwb = aow * stride;
    // B staging: thread t loads row t/4 (0..63), 8 cols at (t&3)*8
    const int ldrB = tid >> 2;
    const int ldcB = (tid & 3) * 8;
    const long wrow = (long)(n0 + ldrB) * Kreal;

    v8f acc00 = {}, acc01 = {}, acc10 = {}, acc11 = {};

    for (int k0 = 0; k0 < Kpad; k0 += 32) {
        // ---- stage A (im2col gather, element granularity) ----
        #pragma unroll
        for (int j = 0; j < 16; ++j) {
            int k = k0 + ldcA + j;
            bf16 vv = (bf16)0.0f;
            if (k < Kreal) {
                int ic = k / KHW; int rem = k - ic * KHW;
                int kh = rem / KW; int kw = rem - kh * KW;
                vv = in[(((long)ab * IC + ic) * IHP + (ihb + kh)) * IWP + (iwb + kw)];
            }
            As[ldrA][ldcA + j] = vv;
        }
        // ---- stage B^T (weights; contiguous 16B per thread) ----
#if HAVE_ASYNC_LDS
        if (asyncB) {
            async_ld16(w + wrow + k0 + ldcB, &Bs[ldrB][ldcB]);
        } else
#endif
        {
            #pragma unroll
            for (int j = 0; j < 8; ++j) {
                int k = k0 + ldcB + j;
                Bs[ldrB][ldcB + j] = (k < Kreal) ? w[wrow + k] : (bf16)0.0f;
            }
        }
        if (k0 + 32 < Kreal)
            __builtin_prefetch(w + wrow + k0 + 32, 0, 1);  // global_prefetch_b8
#if HAVE_ASYNC_LDS
        if (asyncB) __builtin_amdgcn_s_wait_asynccnt(0);
#endif
        __syncthreads();

        // ---- fragments per ISA 7.12.2 layouts ----
        FragB16 a0, a1;
        {
            const uint4* ap0 = (const uint4*)(&As[wm * 16 + lr][0]);
            a0.q[0] = ap0[hi ? 1 : 0];   // K {0..7}  | {8..15}
            a0.q[1] = ap0[hi ? 3 : 2];   // K {16..23}| {24..31}
            const uint4* ap1 = (const uint4*)(&As[64 + wm * 16 + lr][0]);
            a1.q[0] = ap1[hi ? 1 : 0];
            a1.q[1] = ap1[hi ? 3 : 2];
        }
        FragB16 fb0, fb1;
        {
            const uint4* bp0 = (const uint4*)(&Bs[wn * 32 + lr][0]);
            fb0.q[0] = bp0[hi ? 1 : 0]; fb0.q[1] = bp0[hi ? 3 : 2];
            const uint4* bp1 = (const uint4*)(&Bs[wn * 32 + 16 + lr][0]);
            fb1.q[0] = bp1[hi ? 1 : 0]; fb1.q[1] = bp1[hi ? 3 : 2];
        }
        acc00 = __builtin_amdgcn_wmma_f32_16x16x32_bf16(false, a0.v, false, fb0.v,
                                                        (short)0, acc00, false, false);
        acc01 = __builtin_amdgcn_wmma_f32_16x16x32_bf16(false, a0.v, false, fb1.v,
                                                        (short)0, acc01, false, false);
        acc10 = __builtin_amdgcn_wmma_f32_16x16x32_bf16(false, a1.v, false, fb0.v,
                                                        (short)0, acc10, false, false);
        acc11 = __builtin_amdgcn_wmma_f32_16x16x32_bf16(false, a1.v, false, fb1.v,
                                                        (short)0, acc11, false, false);
        __syncthreads();
    }

    // ---- epilogue: bias + relu, scatter to NCHW (padded) bf16 dest ----
    #pragma unroll
    for (int half = 0; half < 2; ++half) {
        v8f aN0 = half ? acc10 : acc00;   // x fb0
        v8f aN1 = half ? acc11 : acc01;   // x fb1
        #pragma unroll
        for (int r = 0; r < 8; ++r) {
            int ml = half * 64 + wm * 16 + (hi ? 8 + r : r);
            int m  = m0 + ml;
            int ow = m % OW; int t = m / OW; int oh = t % OH; int b = t / OH;
            long obase = ((long)b * OC) * DH * DW;
            #pragma unroll
            for (int s2 = 0; s2 < 2; ++s2) {
                int n = n0 + wn * 32 + s2 * 16 + lr;
                float val = (s2 ? aN1[r] : aN0[r]) + bias[n];
                if (relu) val = fmaxf(val, 0.0f);
                out[obase + ((long)n * DH + (oh + padOut)) * DW + (ow + padOut)] = f2b(val);
            }
        }
    }
}

// ------------------------- primary capsule squash --------------------------
// capsule (b,k) = 8 contiguous elements of pc-conv output (576 % 8 == 0)
__global__ void k_primary_squash(const bf16* __restrict__ hpc,
                                 float* __restrict__ u, long nCaps) {
    long i = (long)blockIdx.x * blockDim.x + threadIdx.x;
    if (i >= nCaps) return;
    const bf16* p = hpc + i * 8;
    float t[8]; float nsq = 0.0f;
    #pragma unroll
    for (int a = 0; a < 8; ++a) { t[a] = b2f(p[a]); nsq += t[a] * t[a]; }
    float scale = sqrtf(nsq) / (1.0f + nsq);   // norm_sq/(1+norm_sq) * t/norm
    float* o = u + i * 8;
    #pragma unroll
    for (int a = 0; a < 8; ++a) o[a] = t[a] * scale;
}

// --------------------- u_hat: fp32 WMMA 16x16x4, K=8 -----------------------
// One wave per (j,k): A = W[j,k,:,:] (16d x 8a), B = u[:,k,:]^T (8a x 16b).
__global__ __launch_bounds__(256) void k_uhat(const float* __restrict__ Wc,
                                              const float* __restrict__ u,
                                              float* __restrict__ uhat, int Kc) {
    int g    = (int)((blockIdx.x * 256 + threadIdx.x) >> 5);   // wave id
    int lane = threadIdx.x & 31;
    int lr = lane & 15, hi = lane >> 4;
    int j = g / Kc, k = g - j * Kc;

    const float* wp = Wc + (((long)j * Kc + k) * 16 + lr) * 8;  // row d = lr
    const float* up = u + ((long)lr * Kc + k) * 8;              // col b = lr
    v8f c = {};
#if __has_builtin(__builtin_amdgcn_wmma_f32_16x16x4_f32)
    v2f A0 = { wp[hi * 2 + 0], wp[hi * 2 + 1] };       // K = a {0,1}|{2,3}
    v2f A1 = { wp[4 + hi * 2 + 0], wp[4 + hi * 2 + 1] };
    v2f B0 = { up[hi * 2 + 0], up[hi * 2 + 1] };
    v2f B1 = { up[4 + hi * 2 + 0], up[4 + hi * 2 + 1] };
    c = __builtin_amdgcn_wmma_f32_16x16x4_f32(false, A0, false, B0, (short)0, c, false, false);
    c = __builtin_amdgcn_wmma_f32_16x16x4_f32(false, A1, false, B1, (short)0, c, false, false);
#else
    #pragma unroll
    for (int r = 0; r < 8; ++r) {
        int d = hi ? 8 + r : r;
        const float* wr = Wc + (((long)j * Kc + k) * 16 + d) * 8;
        float sum = 0.0f;
        #pragma unroll
        for (int a = 0; a < 8; ++a) sum += wr[a] * up[a];
        c[r] = sum;
    }
#endif
    float* o = uhat + (((long)lr * 10 + j) * Kc + k) * 16;   // [b][j][k][d]
    #pragma unroll
    for (int r = 0; r < 8; ++r) o[hi ? 8 + r : r] = c[r];
}

// ----------------------------- routing kernels -----------------------------

__global__ void k_softmax_j(const float* __restrict__ bij, float* __restrict__ c, int Kc) {
    long i = (long)blockIdx.x * blockDim.x + threadIdx.x;   // b*Kc + k
    if (i >= 16L * Kc) return;
    int b = (int)(i / Kc); long k = i - (long)b * Kc;
    const float* src = bij + (long)b * 10 * Kc + k;
    float vals[10]; float m = -1e30f;
    #pragma unroll
    for (int j = 0; j < 10; ++j) { vals[j] = src[(long)j * Kc]; m = fmaxf(m, vals[j]); }
    float sum = 0.0f;
    #pragma unroll
    for (int j = 0; j < 10; ++j) { vals[j] = __expf(vals[j] - m); sum += vals[j]; }
    float inv = 1.0f / sum;
    float* dst = c + (long)b * 10 * Kc + k;
    #pragma unroll
    for (int j = 0; j < 10; ++j) dst[(long)j * Kc] = vals[j] * inv;
}

#define SJ_SPLIT 16
// partial s_j over a K-slice: grid = 160 * SJ_SPLIT blocks
__global__ __launch_bounds__(256) void k_sj_part(const float* __restrict__ c,
                                                 const float* __restrict__ uhat,
                                                 float* __restrict__ spart, int Kc) {
    int bj  = blockIdx.x / SJ_SPLIT;
    int sp  = blockIdx.x % SJ_SPLIT;
    int tid = threadIdx.x;
    int kLen = (Kc + SJ_SPLIT - 1) / SJ_SPLIT;
    int kBeg = sp * kLen;
    int kEnd = kBeg + kLen; if (kEnd > Kc) kEnd = Kc;

    float acc[16];
    #pragma unroll
    for (int d = 0; d < 16; ++d) acc[d] = 0.0f;
    const float* cp = c + (long)bj * Kc;
    const float* up = uhat + (long)bj * Kc * 16;
    for (int k = kBeg + tid; k < kEnd; k += 256) {
        float cv = cp[k];
        const float* u16 = up + (long)k * 16;
        #pragma unroll
        for (int d = 0; d < 16; ++d) acc[d] += cv * u16[d];
    }
    __shared__ float red[256];
    for (int d = 0; d < 16; ++d) {
        red[tid] = acc[d]; __syncthreads();
        for (int st = 128; st > 0; st >>= 1) {
            if (tid < st) red[tid] += red[tid + st];
            __syncthreads();
        }
        if (tid == 0) spart[((long)bj * SJ_SPLIT + sp) * 16 + d] = red[0];
        __syncthreads();
    }
}

__global__ void k_sj_final(const float* __restrict__ spart, float* __restrict__ s) {
    int bj = threadIdx.x;
    if (bj >= 160) return;
    #pragma unroll
    for (int d = 0; d < 16; ++d) {
        float sum = 0.0f;
        for (int sp = 0; sp < SJ_SPLIT; ++sp)
            sum += spart[((long)bj * SJ_SPLIT + sp) * 16 + d];
        s[bj * 16 + d] = sum;
    }
}

__global__ void k_squash_digit(const float* __restrict__ s, float* __restrict__ v) {
    int i = threadIdx.x;
    if (i >= 160) return;
    const float* sp = s + i * 16;
    float sq = 0.0f;
    #pragma unroll
    for (int d = 0; d < 16; ++d) sq += sp[d] * sp[d];
    float scale = sq / (1.0f + sq) / sqrtf(sq + 1e-9f);
    float* vp = v + i * 16;
    #pragma unroll
    for (int d = 0; d < 16; ++d) vp[d] = sp[d] * scale;
}

__global__ void k_bupdate(const float* __restrict__ uhat, const float* __restrict__ v,
                          float* __restrict__ bij, int Kc) {
    long i = (long)blockIdx.x * blockDim.x + threadIdx.x;   // (b*10+j)*Kc + k
    if (i >= 160L * Kc) return;
    long bj = i / Kc;
    const float* up = uhat + i * 16;
    const float* vp = v + bj * 16;
    float dot = 0.0f;
    #pragma unroll
    for (int d = 0; d < 16; ++d) dot += up[d] * vp[d];
    bij[i] += dot;
}

// ------------------ capsule lengths, argmax one-hot mask -------------------
__global__ void k_len_mask(const float* __restrict__ v, float* __restrict__ out_caps,
                           bf16* __restrict__ masked) {
    int b = threadIdx.x;
    if (b >= 16) return;
    int amax = 0; float best = -1.0f;
    for (int j = 0; j < 10; ++j) {
        const float* vp = v + (b * 10 + j) * 16;
        float sq = 1e-9f;
        #pragma unroll
        for (int d = 0; d < 16; ++d) sq += vp[d] * vp[d];
        float len = sqrtf(sq);
        out_caps[b * 10 + j] = len;
        if (len > best) { best = len; amax = j; }
    }
    for (int j = 0; j < 10; ++j)
        #pragma unroll
        for (int d = 0; d < 16; ++d)
            masked[b * 160 + j * 16 + d] =
                (j == amax) ? f2b(v[(b * 10 + j) * 16 + d]) : (bf16)0.0f;
}

// -------------------- decoder FC (bf16 WMMA, M = 16) -----------------------
// act: [16,K] bf16, wT: [N,K] bf16 (Linear layout == B^T already),
// one wave per 16-column tile of the output.
__global__ __launch_bounds__(256) void k_fc_wmma(
    const bf16* __restrict__ act, const bf16* __restrict__ wT,
    const float* __restrict__ bias, bf16* __restrict__ outb,
    float* __restrict__ outf, int K, int N, int mode)
{
    int wvg  = (int)((blockIdx.x * 256 + threadIdx.x) >> 5);
    int lane = threadIdx.x & 31;
    int lr = lane & 15, hi = lane >> 4;
    int n0 = wvg * 16;

    v8f acc = {};
    for (int k0 = 0; k0 < K; k0 += 32) {
        FragB16 a, bb;
        const uint4* ap = (const uint4*)(act + (long)lr * K + k0);
        a.q[0] = ap[hi ? 1 : 0]; a.q[1] = ap[hi ? 3 : 2];
        const uint4* bp = (const uint4*)(wT + (long)(n0 + lr) * K + k0);
        bb.q[0] = bp[hi ? 1 : 0]; bb.q[1] = bp[hi ? 3 : 2];
        acc = __builtin_amdgcn_wmma_f32_16x16x32_bf16(false, a.v, false, bb.v,
                                                      (short)0, acc, false, false);
    }
    #pragma unroll
    for (int r = 0; r < 8; ++r) {
        int m = hi ? 8 + r : r;
        int n = n0 + lr;
        float val = acc[r] + bias[n];
        if (mode == 1) val = fmaxf(val, 0.0f);
        else if (mode == 2) val = 1.0f / (1.0f + __expf(-val));
        if (outb) outb[(long)m * N + n] = f2b(val);
        if (outf) outf[(long)m * N + n] = val;
    }
}

// ------------------------------- host side ---------------------------------

extern "C" void kernel_launch(void* const* d_in, const int* in_sizes, int n_in,
                              void* d_out, int out_size, void* d_ws, size_t ws_size,
                              hipStream_t stream) {
    (void)in_sizes; (void)n_in; (void)out_size; (void)ws_size;
    const float* x    = (const float*)d_in[0];
    const float* c1w  = (const float*)d_in[1];
    const float* c1b  = (const float*)d_in[2];
    const float* c2w  = (const float*)d_in[3];
    const float* c2b  = (const float*)d_in[4];
    const float* pcw  = (const float*)d_in[5];
    const float* pcb  = (const float*)d_in[6];
    const float* Wc   = (const float*)d_in[7];
    const float* f1w  = (const float*)d_in[8];
    const float* f1b  = (const float*)d_in[9];
    const float* f2w  = (const float*)d_in[10];
    const float* f2b  = (const float*)d_in[11];
    const float* f3w  = (const float*)d_in[12];
    const float* f3b  = (const float*)d_in[13];
    float* out = (float*)d_out;

    const int Kc = 18432;

    // bump allocator over d_ws (~320 MB total)
    char* p = (char*)d_ws;
    auto alloc = [&](size_t bytes) -> char* {
        char* r = p; p += (bytes + 255) & ~(size_t)255; return r;
    };
    bf16*  x_bf   = (bf16*)alloc((size_t)16 * 64 * 64 * 2);
    bf16*  w1_bf  = (bf16*)alloc((size_t)256 * 81 * 2);
    bf16*  w2_bf  = (bf16*)alloc((size_t)256 * 256 * 25 * 2);
    bf16*  wpc_bf = (bf16*)alloc((size_t)256 * 256 * 81 * 2);
    bf16*  f1_bf  = (bf16*)alloc((size_t)512 * 160 * 2);
    bf16*  f2_bf  = (bf16*)alloc((size_t)1024 * 512 * 2);
    bf16*  f3_bf  = (bf16*)alloc((size_t)4096 * 1024 * 2);
    bf16*  h1     = (bf16*)alloc((size_t)16 * 256 * 60 * 60 * 2);  // padded for conv2
    bf16*  h2     = (bf16*)alloc((size_t)16 * 256 * 56 * 56 * 2);
    bf16*  hpc    = (bf16*)alloc((size_t)16 * 256 * 24 * 24 * 2);
    float* u      = (float*)alloc((size_t)16 * Kc * 8 * 4);
    float* uhat   = (float*)alloc((size_t)16 * 10 * Kc * 16 * 4);
    float* bij    = (float*)alloc((size_t)16 * 10 * Kc * 4);
    float* cc     = (float*)alloc((size_t)16 * 10 * Kc * 4);
    float* spart  = (float*)alloc((size_t)160 * SJ_SPLIT * 16 * 4);
    float* s      = (float*)alloc(2560 * 4);
    float* v      = (float*)alloc(2560 * 4);
    bf16*  masked = (bf16*)alloc(2560 * 2);
    bf16*  d1     = (bf16*)alloc((size_t)16 * 512 * 2);
    bf16*  d2     = (bf16*)alloc((size_t)16 * 1024 * 2);

    auto cvt = [&](const float* src, bf16* dst, long n) {
        long b = (n + 255) / 256; if (b > 4096) b = 4096;
        k_cvt_bf16<<<dim3((unsigned)b), dim3(256), 0, stream>>>(src, dst, n);
    };
    cvt(x,   x_bf,   16L * 64 * 64);
    cvt(c1w, w1_bf,  256L * 81);
    cvt(c2w, w2_bf,  256L * 256 * 25);
    cvt(pcw, wpc_bf, 256L * 256 * 81);
    cvt(f1w, f1_bf,  512L * 160);
    cvt(f2w, f2_bf,  1024L * 512);
    cvt(f3w, f3_bf,  4096L * 1024);

    // zero padded conv2-input ring + routing logits
    k_zero_u32<<<dim3(4096), dim3(256), 0, stream>>>((uint32_t*)h1, (16L * 256 * 60 * 60 * 2) / 4);
    k_zero_u32<<<dim3(4096), dim3(256), 0, stream>>>((uint32_t*)bij, 16L * 10 * Kc);

    // conv1: [16,1,64,64] -> [16,256,56,56], relu, into padded 60x60 dest
    k_conv_wmma<<<dim3(392 * 4), dim3(256), 0, stream>>>(
        x_bf, w1_bf, c1b, h1, 16, 1, 64, 64, 256, 56, 56, 9, 9, 1,
        60, 60, 2, 81, 1);
    // conv2: padded 60x60 -> [16,256,56,56], relu
    k_conv_wmma<<<dim3(392 * 4), dim3(256), 0, stream>>>(
        h1, w2_bf, c2b, h2, 16, 256, 60, 60, 256, 56, 56, 5, 5, 1,
        56, 56, 0, 256 * 25, 1);
    // primary caps conv: stride 2 -> [16,256,24,24]
    k_conv_wmma<<<dim3(72 * 4), dim3(256), 0, stream>>>(
        h2, wpc_bf, pcb, hpc, 16, 256, 56, 56, 256, 24, 24, 9, 9, 2,
        24, 24, 0, 256 * 81, 0);

    // primary squash: 16*18432 capsules
    k_primary_squash<<<dim3(1152), dim3(256), 0, stream>>>(hpc, u, 16L * Kc);

    // u_hat: one wave per (j,k); 10*18432 waves / 8 per block
    k_uhat<<<dim3(23040), dim3(256), 0, stream>>>(Wc, u, uhat, Kc);

    // dynamic routing, 3 iterations
    for (int it = 0; it < 3; ++it) {
        k_softmax_j<<<dim3(1152), dim3(256), 0, stream>>>(bij, cc, Kc);
        k_sj_part<<<dim3(160 * SJ_SPLIT), dim3(256), 0, stream>>>(cc, uhat, spart, Kc);
        k_sj_final<<<dim3(1), dim3(256), 0, stream>>>(spart, s);
        k_squash_digit<<<dim3(1), dim3(256), 0, stream>>>(s, v);
        if (it < 2)
            k_bupdate<<<dim3(11520), dim3(256), 0, stream>>>(uhat, v, bij, Kc);
    }

    // capsule lengths -> d_out[0:160], argmax mask -> bf16 [16,160]
    k_len_mask<<<dim3(1), dim3(32), 0, stream>>>(v, out, masked);

    // decoder: 160 -> 512 -> 1024 -> 4096 (sigmoid), recon -> d_out + 160
    k_fc_wmma<<<dim3(4),  dim3(256), 0, stream>>>(masked, f1_bf, f1b, d1, (float*)nullptr, 160,  512,  1);
    k_fc_wmma<<<dim3(8),  dim3(256), 0, stream>>>(d1,     f2_bf, f2b, d2, (float*)nullptr, 512,  1024, 1);
    k_fc_wmma<<<dim3(32), dim3(256), 0, stream>>>(d2,     f3_bf, f3b, (bf16*)nullptr, out + 160, 1024, 4096, 2);
}
